// DeformableMultiHeadAttention_38491496907031
// MI455X (gfx1250) — matrix operand order repaired and checked by reading, not compiled, not run on hardware
//
#include <hip/hip_runtime.h>
#include <hip/hip_bf16.h>
#include <math.h>

typedef __attribute__((ext_vector_type(16))) _Float16 v16h;
typedef __attribute__((ext_vector_type(8)))  _Float16 v8h;
typedef __attribute__((ext_vector_type(8)))  float    v8f;

#define HW64   64
#define NTOK   4096      // H*W
#define DMODEL 256
#define NH     8
#define HD     32
#define NPNT   4
#define KDIM   256

union Frag16 { v16h v; v8h h[2]; };

__device__ __forceinline__ v8h load8h(const _Float16* p) { return *(const v8h*)p; }

// ---------------- convert x (f32) -> f16 ----------------
__global__ void k_cvt_f16(const float* __restrict__ src, _Float16* __restrict__ dst, int count) {
  int i = blockIdx.x * blockDim.x + threadIdx.x;
  if (i < count) dst[i] = (_Float16)src[i];
}

// ---------------- pack weight: WT[n*K + k] = W[k*srcld + col0 + n] ----------------
__global__ void k_pack_w(const float* __restrict__ W, _Float16* __restrict__ WT,
                         int Nc, int col0, int srcld) {
  int i = blockIdx.x * blockDim.x + threadIdx.x;
  if (i >= Nc * KDIM) return;
  int n = i >> 8;          // i / 256
  int k = i & (KDIM - 1);  // i % 256
  WT[i] = (_Float16)W[k * srcld + col0 + n];
}

// ---------------- WMMA 32x32 tile core (K = 256, f16 in, f32 acc) ----------------
__device__ __forceinline__ void wmma_tile_32x32(const _Float16* __restrict__ A,
                                                const _Float16* __restrict__ BT,
                                                int m0, int n0, int lane,
                                                v8f acc[2][2]) {
  const int r      = lane & 15;
  const int kselA  = (lane >> 4) << 4;   // A: 0 or 16 (one contiguous 16-half chunk)
  const int kselB  = (lane >> 4) << 4;   // B: 0 or 16 (one contiguous 16-half chunk)
  const _Float16* pa0 = A  + (size_t)(m0 + r)      * KDIM + kselA;
  const _Float16* pa1 = pa0 + (size_t)16 * KDIM;
  const _Float16* pb0 = BT + (size_t)(n0 + r)      * KDIM + kselB;
  const _Float16* pb1 = pb0 + (size_t)16 * KDIM;
  #pragma unroll
  for (int k0 = 0; k0 < KDIM; k0 += 32) {
    Frag16 a0, a1, b0, b1;
    a0.h[0] = load8h(pa0 + k0);     a0.h[1] = load8h(pa0 + k0 + 8);
    a1.h[0] = load8h(pa1 + k0);     a1.h[1] = load8h(pa1 + k0 + 8);
    b0.h[0] = load8h(pb0 + k0);     b0.h[1] = load8h(pb0 + k0 + 8);
    b1.h[0] = load8h(pb1 + k0);     b1.h[1] = load8h(pb1 + k0 + 8);
    acc[0][0] = __builtin_amdgcn_wmma_f32_16x16x32_f16(false, a0.v, false, b0.v, (short)0, acc[0][0], false, false);
    acc[0][1] = __builtin_amdgcn_wmma_f32_16x16x32_f16(false, a0.v, false, b1.v, (short)0, acc[0][1], false, false);
    acc[1][0] = __builtin_amdgcn_wmma_f32_16x16x32_f16(false, a1.v, false, b0.v, (short)0, acc[1][0], false, false);
    acc[1][1] = __builtin_amdgcn_wmma_f32_16x16x32_f16(false, a1.v, false, b1.v, (short)0, acc[1][1], false, false);
  }
}

// ---------------- GEMM -> plain row-major f32 [M][Nc] with bias ----------------
__global__ __launch_bounds__(256) void k_gemm_plain(const _Float16* __restrict__ A,
                                                    const _Float16* __restrict__ BT,
                                                    const float* __restrict__ bias,
                                                    float* __restrict__ C,
                                                    int M, int Nc) {
  const int lane = threadIdx.x & 31;
  const int gw   = blockIdx.x * (blockDim.x >> 5) + (threadIdx.x >> 5);
  const int tilesN = Nc >> 5;
  const int total  = (M >> 5) * tilesN;
  if (gw >= total) return;                  // wave-uniform: EXEC stays all-1s
  const int m0 = (gw / tilesN) << 5;
  const int n0 = (gw % tilesN) << 5;
  v8f acc[2][2] = {};
  wmma_tile_32x32(A, BT, m0, n0, lane, acc);
  const int mbase = m0 + ((lane >> 4) << 3);
  const int nlow  = lane & 15;
  #pragma unroll
  for (int mi = 0; mi < 2; ++mi)
    #pragma unroll
    for (int ni = 0; ni < 2; ++ni) {
      const int n  = n0 + ni * 16 + nlow;
      const float bv = bias[n];
      #pragma unroll
      for (int rr = 0; rr < 8; ++rr) {
        const int m = mbase + mi * 16 + rr;
        C[(size_t)m * Nc + n] = acc[mi][ni][rr] + bv;
      }
    }
}

// ---------------- GEMM -> v laid out as [B][NH][NTOK][HD] f32, with bias ----------------
__global__ __launch_bounds__(256) void k_gemm_vperm(const _Float16* __restrict__ A,
                                                    const _Float16* __restrict__ BT,
                                                    const float* __restrict__ bias,
                                                    float* __restrict__ V, int M) {
  const int lane = threadIdx.x & 31;
  const int gw   = blockIdx.x * (blockDim.x >> 5) + (threadIdx.x >> 5);
  const int tilesN = DMODEL >> 5;           // 8
  const int total  = (M >> 5) * tilesN;
  if (gw >= total) return;
  const int m0 = (gw / tilesN) << 5;
  const int n0 = (gw % tilesN) << 5;
  v8f acc[2][2] = {};
  wmma_tile_32x32(A, BT, m0, n0, lane, acc);
  const int mbase = m0 + ((lane >> 4) << 3);
  const int nlow  = lane & 15;
  #pragma unroll
  for (int mi = 0; mi < 2; ++mi)
    #pragma unroll
    for (int ni = 0; ni < 2; ++ni) {
      const int nn = n0 + ni * 16 + nlow;   // model dim d
      const int h  = nn >> 5, c = nn & 31;
      const float bv = bias[nn];
      #pragma unroll
      for (int rr = 0; rr < 8; ++rr) {
        const int m   = mbase + mi * 16 + rr;   // global token row
        const int bi  = m >> 12;                // / NTOK
        const int tok = m & (NTOK - 1);
        V[(((size_t)(bi * NH + h)) * NTOK + tok) * HD + c] = acc[mi][ni][rr] + bv;
      }
    }
}

// ---------------- offsets -> pixel coords (in place), softmax over P (in place) ----------------
__global__ void k_loc_attn(float* __restrict__ offp,   // [M][NH][NPNT][2]
                           float* __restrict__ attp,   // [M][NH][NPNT]
                           int M) {
  int i = blockIdx.x * blockDim.x + threadIdx.x;       // over M*NH
  if (i >= M * NH) return;
  const int t   = i >> 3;
  const int h   = i & 7;
  const int tok = t & (NTOK - 1);
  const float rx = (float)(tok & (HW64 - 1));
  const float ry = (float)(tok >> 6);
  float* op = offp + (size_t)t * (NH * NPNT * 2) + h * (NPNT * 2);
  #pragma unroll
  for (int p = 0; p < NPNT; ++p) {
    op[2 * p]     = rx + 2.0f * tanhf(op[2 * p]);
    op[2 * p + 1] = ry + 2.0f * tanhf(op[2 * p + 1]);
  }
  float* ap = attp + (size_t)t * (NH * NPNT) + h * NPNT;
  float a0 = ap[0], a1 = ap[1], a2 = ap[2], a3 = ap[3];
  float mx = fmaxf(fmaxf(a0, a1), fmaxf(a2, a3));
  float e0 = expf(a0 - mx), e1 = expf(a1 - mx), e2 = expf(a2 - mx), e3 = expf(a3 - mx);
  float inv = 1.0f / (e0 + e1 + e2 + e3);
  ap[0] = e0 * inv; ap[1] = e1 * inv; ap[2] = e2 * inv; ap[3] = e3 * inv;
}

// ---------------- bilinear sample + attn-weighted sum -> mid f16 [M][DMODEL] ----------------
__global__ __launch_bounds__(256) void k_sample(const float* __restrict__ V,
                                                const float* __restrict__ loc,
                                                const float* __restrict__ attn,
                                                _Float16* __restrict__ mid) {
  const int t = blockIdx.x;          // token = b*NTOK + n
  const int d = threadIdx.x;         // 0..255
  const int h = d >> 5, c = d & 31;
  const int bi = t >> 12;
  const float* lp = loc  + (size_t)t * (NH * NPNT * 2) + h * (NPNT * 2);
  const float* ap = attn + (size_t)t * (NH * NPNT)     + h * NPNT;
  const float* vb = V + (((size_t)(bi * NH + h)) * NTOK) * HD + c;
  float sum = 0.f;
  #pragma unroll
  for (int p = 0; p < NPNT; ++p) {
    const float gx = lp[2 * p], gy = lp[2 * p + 1];
    const float x0f = floorf(gx), y0f = floorf(gy);
    const int   x0  = (int)x0f,   y0  = (int)y0f;
    const float wx1 = gx - x0f, wx0 = 1.f - wx1;
    const float wy1 = gy - y0f, wy0 = 1.f - wy1;
    float bil = 0.f;
    #pragma unroll
    for (int cy = 0; cy < 2; ++cy) {
      const int yi = y0 + cy;
      if (yi < 0 || yi >= HW64) continue;
      const float wy = cy ? wy1 : wy0;
      #pragma unroll
      for (int cx = 0; cx < 2; ++cx) {
        const int xi = x0 + cx;
        if (xi < 0 || xi >= HW64) continue;
        bil += ((cx ? wx1 : wx0) * wy) * vb[(size_t)(yi * HW64 + xi) * HD];
      }
    }
    sum += ap[p] * bil;
  }
  mid[(size_t)t * DMODEL + d] = (_Float16)sum;
}

// ---------------- host launcher ----------------
extern "C" void kernel_launch(void* const* d_in, const int* in_sizes, int n_in,
                              void* d_out, int out_size, void* d_ws, size_t ws_size,
                              hipStream_t stream) {
  (void)n_in; (void)out_size; (void)ws_size;
  const float* x      = (const float*)d_in[0];
  const float* qkv_w  = (const float*)d_in[1];
  const float* qkv_b  = (const float*)d_in[2];
  const float* off_w  = (const float*)d_in[3];
  const float* off_b  = (const float*)d_in[4];
  const float* attw_w = (const float*)d_in[5];
  const float* attw_b = (const float*)d_in[6];
  const float* proj_w = (const float*)d_in[7];
  const float* proj_b = (const float*)d_in[8];
  float* out = (float*)d_out;

  const int M = in_sizes[0] / DMODEL;     // B * N tokens

  char* p = (char*)d_ws;
  auto carve = [&](size_t bytes) -> char* {
    char* r = p; p += (bytes + 255) & ~(size_t)255; return r;
  };
  _Float16* xh    = (_Float16*)carve((size_t)M * KDIM * 2);
  _Float16* wtv   = (_Float16*)carve((size_t)DMODEL * KDIM * 2);
  _Float16* wtoff = (_Float16*)carve((size_t)(NH * NPNT * 2) * KDIM * 2);
  _Float16* wtatt = (_Float16*)carve((size_t)(NH * NPNT) * KDIM * 2);
  _Float16* wtprj = (_Float16*)carve((size_t)DMODEL * KDIM * 2);
  float*    vbuf  = (float*)   carve((size_t)M * DMODEL * 4);
  float*    offp  = (float*)   carve((size_t)M * (NH * NPNT * 2) * 4);
  float*    attp  = (float*)   carve((size_t)M * (NH * NPNT) * 4);
  _Float16* mid   = (_Float16*)carve((size_t)M * DMODEL * 2);

  const int cvtN = M * KDIM;
  k_cvt_f16<<<(cvtN + 255) / 256, 256, 0, stream>>>(x, xh, cvtN);
  k_pack_w<<<(DMODEL * KDIM + 255) / 256, 256, 0, stream>>>(qkv_w, wtv, DMODEL, 2 * DMODEL, 3 * DMODEL);
  k_pack_w<<<(64 * KDIM + 255) / 256, 256, 0, stream>>>(off_w, wtoff, 64, 0, 64);
  k_pack_w<<<(32 * KDIM + 255) / 256, 256, 0, stream>>>(attw_w, wtatt, 32, 0, 32);
  k_pack_w<<<(DMODEL * KDIM + 255) / 256, 256, 0, stream>>>(proj_w, wtprj, DMODEL, 0, DMODEL);

  const int tilesV = (M >> 5) * (DMODEL >> 5);
  k_gemm_vperm<<<(tilesV + 7) / 8, 256, 0, stream>>>(xh, wtv, qkv_b + 2 * DMODEL, vbuf, M);
  const int tilesO = (M >> 5) * (64 >> 5);
  k_gemm_plain<<<(tilesO + 7) / 8, 256, 0, stream>>>(xh, wtoff, off_b, offp, M, 64);
  const int tilesA = (M >> 5) * (32 >> 5);
  k_gemm_plain<<<(tilesA + 7) / 8, 256, 0, stream>>>(xh, wtatt, attw_b, attp, M, 32);

  k_loc_attn<<<(M * NH + 255) / 256, 256, 0, stream>>>(offp, attp, M);
  k_sample<<<M, 256, 0, stream>>>(vbuf, offp, attp, mid);

  const int tilesP = (M >> 5) * (DMODEL >> 5);
  k_gemm_plain<<<(tilesP + 7) / 8, 256, 0, stream>>>(mid, wtprj, proj_b, out, M, DMODEL);
}
